// StrongGNNMultiLabelCardinality_2310692405730
// MI455X (gfx1250) — matrix operand order repaired and checked by reading, not compiled, not run on hardware
//
#include <hip/hip_runtime.h>
#include <math.h>

typedef __attribute__((ext_vector_type(16))) __bf16 v16bf;
typedef __attribute__((ext_vector_type(8)))  float  v8f;

struct __align__(16) U4 { unsigned int x, y, z, w; };

__device__ __forceinline__ unsigned short f2bf(float f) {
  unsigned u = __float_as_uint(f);
  u += 0x7FFFu + ((u >> 16) & 1u);           // round-to-nearest-even
  return (unsigned short)(u >> 16);
}

__device__ __forceinline__ void atomicMaxF(float* addr, float v) {
  if (v >= 0.0f) atomicMax((int*)addr, __float_as_int(v));
  else           atomicMin((unsigned int*)addr, __float_as_uint(v));
}

// ---------------------------------------------------------------------------
// Generic bf16 WMMA GEMM:  C[M,N] = act(A[M,K] @ Wt[N,K]^T + bias)
//   - optional gather A-stage (msg1):  A-row r = [ h_bf16[src[r]] , bf16(edge_attr[r]) , 0pad ]
//   - optional scatter epilogue (msg2): atomicAdd into Cscat[scatIdx[r]*N + col]
// Block = 256 threads (8 wave32 waves), tile 64(M) x 128(N), K step 32.
// Double-buffered LDS: one barrier per K-step, global loads overlap WMMA.
// ---------------------------------------------------------------------------
#define TM 64
#define TN 128
#define LDT 40   // LDS row stride in shorts (32 k-values + pad), 80B = 16B aligned

__device__ __forceinline__ U4 stageA_val(const unsigned short* __restrict__ A,
                                         const int* __restrict__ gatherSrc,
                                         const unsigned short* __restrict__ Hsrc,
                                         const float* __restrict__ EA,
                                         int M, int K, int m0, int t, int k0) {
  int row = t >> 2;
  int col = (t & 3) * 8;
  int gr  = m0 + row;
  U4 val = {0u, 0u, 0u, 0u};
  if (gr < M) {
    int gk = k0 + col;
    if (gatherSrc == nullptr) {
      val = *(const U4*)(A + (size_t)gr * K + gk);
    } else {
      if (gk < 256) {
        int s = gatherSrc[gr];
        val = *(const U4*)(Hsrc + (size_t)s * 256 + gk);
      } else if (gk < 272) {
        const float* ep = EA + (size_t)gr * 16 + (gk - 256);
        float4 f0 = *(const float4*)(ep);
        float4 f1 = *(const float4*)(ep + 4);
        alignas(16) unsigned short hh[8];
        hh[0]=f2bf(f0.x); hh[1]=f2bf(f0.y); hh[2]=f2bf(f0.z); hh[3]=f2bf(f0.w);
        hh[4]=f2bf(f1.x); hh[5]=f2bf(f1.y); hh[6]=f2bf(f1.z); hh[7]=f2bf(f1.w);
        __builtin_memcpy(&val, hh, 16);
      } // else zero pad (K 272..287)
    }
  }
  return val;
}

__device__ __forceinline__ void stageW_val(const unsigned short* __restrict__ Wt,
                                           int K, int n0, int t, int k0,
                                           U4& w0, U4& w1) {
  int n   = t >> 1;
  int col = (t & 1) * 16;
  const unsigned short* wp = Wt + (size_t)(n0 + n) * K + k0 + col;
  w0 = *(const U4*)(wp);
  w1 = *(const U4*)(wp + 8);
}

__global__ __launch_bounds__(256) void k_gemm(
    const unsigned short* __restrict__ A,
    const unsigned short* __restrict__ Wt,
    const float* __restrict__ bias,
    float* __restrict__ Cf,
    unsigned short* __restrict__ Cb,
    float* __restrict__ Cscat,
    const int* __restrict__ scatIdx,
    const int* __restrict__ gatherSrc,
    const unsigned short* __restrict__ Hsrc,   // h bf16 [*,256]
    const float* __restrict__ EA,              // edge_attr f32 [*,16]
    int M, int N, int K, int act)
{
  __shared__ unsigned short Asm[2][TM * LDT];
  __shared__ unsigned short Wsm[2][TN * LDT];

  const int t    = threadIdx.x;
  const int lane = t & 31;
  const int w    = t >> 5;
  const int m0   = blockIdx.x * TM;
  const int n0   = blockIdx.y * TN;

  v8f acc[4];
  for (int s = 0; s < 4; ++s)
    for (int i = 0; i < 8; ++i) acc[s][i] = 0.0f;

  const int mloc  = (w & 3) * 16 + (lane & 15);  // A-frag row in tile
  const int ak0   = (lane >> 4) * 8;             // A-frag k base (doc layout)
  const int bbase = (lane >> 4) * 8;             // B-frag k base
  const int nsub0 = (w >> 2) * 64;               // wave's n-subgroup
  const int arow  = t >> 2, acol = (t & 3) * 8;  // staging coords
  const int wrow  = t >> 1, wcol = (t & 1) * 16;

  // ---- stage tile 0 ----
  {
    U4 a0 = stageA_val(A, gatherSrc, Hsrc, EA, M, K, m0, t, 0);
    U4 w0, w1; stageW_val(Wt, K, n0, t, 0, w0, w1);
    *(U4*)(&Asm[0][arow * LDT + acol])     = a0;
    *(U4*)(&Wsm[0][wrow * LDT + wcol])     = w0;
    *(U4*)(&Wsm[0][wrow * LDT + wcol + 8]) = w1;
  }
  __syncthreads();

  const int nt = K >> 5;
  for (int kt = 0; kt < nt; ++kt) {
    const int cur = kt & 1;
    // issue next tile's global loads early (overlap with WMMA)
    U4 aN = {0u,0u,0u,0u}, wN0 = {0u,0u,0u,0u}, wN1 = {0u,0u,0u,0u};
    const bool more = (kt + 1) < nt;
    if (more) {
      aN = stageA_val(A, gatherSrc, Hsrc, EA, M, K, m0, t, (kt + 1) << 5);
      stageW_val(Wt, K, n0, t, (kt + 1) << 5, wN0, wN1);
    }

    // ---- fragments & WMMA from current buffer ----
    const unsigned short* Ab = Asm[cur];
    const unsigned short* Wb = Wsm[cur];
    alignas(16) U4 abits[2];
    abits[0] = *(const U4*)(Ab + mloc * LDT + ak0);
    abits[1] = *(const U4*)(Ab + mloc * LDT + ak0 + 16);
    v16bf av;
    __builtin_memcpy(&av, abits, 32);
#pragma unroll
    for (int s = 0; s < 4; ++s) {
      int nl = nsub0 + s * 16 + (lane & 15);
      alignas(16) U4 bbits[2];
      bbits[0] = *(const U4*)(Wb + nl * LDT + bbase);
      bbits[1] = *(const U4*)(Wb + nl * LDT + bbase + 16);
      v16bf bv;
      __builtin_memcpy(&bv, bbits, 32);
      acc[s] = __builtin_amdgcn_wmma_f32_16x16x32_bf16(
          false, av, false, bv, (short)0, acc[s], false, false);
    }

    // ---- write next tile into the other buffer ----
    if (more) {
      const int nxt = cur ^ 1;
      *(U4*)(&Asm[nxt][arow * LDT + acol])     = aN;
      *(U4*)(&Wsm[nxt][wrow * LDT + wcol])     = wN0;
      *(U4*)(&Wsm[nxt][wrow * LDT + wcol + 8]) = wN1;
    }
    __syncthreads();
  }

  // ---- epilogue: bias + activation + (store f32 / bf16 | scatter-atomic) ----
  const int mrow_base = (w & 3) * 16 + (lane >> 4) * 8;
#pragma unroll
  for (int s = 0; s < 4; ++s) {
    int gc = n0 + nsub0 + s * 16 + (lane & 15);
    float bv = bias ? bias[gc] : 0.0f;
#pragma unroll
    for (int i = 0; i < 8; ++i) {
      int gr = m0 + mrow_base + i;
      if (gr >= M) continue;
      float v = acc[s][i] + bv;
      if (act == 1)      v = fmaxf(v, 0.0f);
      else if (act == 2) v = 1.0f / (1.0f + __expf(-v));
      if (Cscat) atomicAdd(Cscat + (size_t)scatIdx[gr] * N + gc, v);
      if (Cf)    Cf[(size_t)gr * N + gc] = v;
      if (Cb)    Cb[(size_t)gr * N + gc] = f2bf(v);
    }
  }
}

// ---------------------------------------------------------------------------
// Support kernels
// ---------------------------------------------------------------------------
__global__ void k_cvt(const float* __restrict__ s, unsigned short* __restrict__ d, size_t n) {
  size_t i = (size_t)blockIdx.x * 256 + threadIdx.x;
  if (i < n) d[i] = f2bf(s[i]);
}

__global__ void k_cvt_pad(const float* __restrict__ s, unsigned short* __restrict__ d,
                          int I, int Ip, int total) {
  int i = blockIdx.x * 256 + threadIdx.x;
  if (i >= total) return;
  int o = i / Ip, c = i % Ip;
  d[i] = (c < I) ? f2bf(s[(size_t)o * I + c]) : (unsigned short)0;
}

__global__ void k_zero(float* __restrict__ p, size_t n) {
  size_t i = (size_t)blockIdx.x * 256 + threadIdx.x;
  if (i < n) p[i] = 0.0f;
}

__global__ void k_gru(const float* __restrict__ gi, const float* __restrict__ gh,
                      float* __restrict__ h, unsigned short* __restrict__ hb,
                      const float* __restrict__ bng, const float* __restrict__ bnb) {
  int n = blockIdx.x, c = threadIdx.x;
  size_t b3 = (size_t)n * 768;
  float ir = gi[b3 + c], iz = gi[b3 + 256 + c], inn = gi[b3 + 512 + c];
  float hr = gh[b3 + c], hz = gh[b3 + 256 + c], hn = gh[b3 + 512 + c];
  float r  = 1.0f / (1.0f + __expf(-(ir + hr)));
  float z  = 1.0f / (1.0f + __expf(-(iz + hz)));
  float nn = tanhf(inn + r * hn);
  size_t ix = (size_t)n * 256 + c;
  float hold = h[ix];
  float hnew = (1.0f - z) * nn + z * hold;
  float out  = hold + (bng[c] * hnew * rsqrtf(1.0f + 1e-5f) + bnb[c]);
  h[ix] = out; hb[ix] = f2bf(out);
}

__global__ void k_pool_init(float* zs, float* zm, float* cnt, int G) {
  int i = blockIdx.x * 256 + threadIdx.x;
  if (i < G * 256) { zs[i] = 0.0f; zm[i] = -3.402823466e38f; }
  if (i < G) cnt[i] = 0.0f;
}

__global__ void k_pool(const float* __restrict__ h, const int* __restrict__ batch,
                       float* zs, float* zm, float* cnt) {
  int n = blockIdx.x, c = threadIdx.x;
  int g = batch[n];
  float v = h[(size_t)n * 256 + c];
  atomicAdd(zs + (size_t)g * 256 + c, v);
  atomicMaxF(zm + (size_t)g * 256 + c, v);
  if (c == 0) atomicAdd(cnt + g, 1.0f);
}

__global__ void k_pool_final(const float* zs, const float* zm, const float* cnt,
                             unsigned short* zcat, int G) {
  int i = blockIdx.x * 256 + threadIdx.x;
  if (i >= G * 256) return;
  int g = i >> 8, c = i & 255;
  float ct = cnt[g];
  float mean = zs[i] / fmaxf(ct, 1.0f);
  float mx   = (ct > 0.0f) ? zm[i] : 0.0f;
  zcat[(size_t)g * 512 + c]       = f2bf(mean);
  zcat[(size_t)g * 512 + 256 + c] = f2bf(mx);
}

__global__ void k_concat2(const float* __restrict__ a, const float* __restrict__ b,
                          unsigned short* __restrict__ o, int wa, int wb, int rows) {
  int wt = wa + wb;
  int i = blockIdx.x * 256 + threadIdx.x;
  if (i >= rows * wt) return;
  int r = i / wt, c = i % wt;
  float v = (c < wa) ? a[(size_t)r * wa + c] : b[(size_t)r * wb + (c - wa)];
  o[i] = f2bf(v);
}

// mode 0: y = g*a + (1-g)*b     mode 1: y = a + g*b
__global__ void k_mix(const float* __restrict__ g, const float* __restrict__ a,
                      const float* __restrict__ b, float* __restrict__ y,
                      int total, int mode) {
  int i = blockIdx.x * 256 + threadIdx.x;
  if (i >= total) return;
  float gv = g[i];
  y[i] = (mode == 0) ? (gv * a[i] + (1.0f - gv) * b[i]) : (a[i] + gv * b[i]);
}

__global__ void k_ln(const float* __restrict__ x, const float* __restrict__ gam,
                     const float* __restrict__ bet, float* __restrict__ y) {
  __shared__ float red[256];
  int r = blockIdx.x, c = threadIdx.x;
  float v = x[(size_t)r * 256 + c];
  red[c] = v; __syncthreads();
  for (int s = 128; s > 0; s >>= 1) { if (c < s) red[c] += red[c + s]; __syncthreads(); }
  float mu = red[0] / 256.0f; __syncthreads();
  float d = v - mu;
  red[c] = d * d; __syncthreads();
  for (int s = 128; s > 0; s >>= 1) { if (c < s) red[c] += red[c + s]; __syncthreads(); }
  float var = red[0] / 256.0f;
  y[(size_t)r * 256 + c] = d * rsqrtf(var + 1e-5f) * gam[c] + bet[c];
}

__global__ void k_feat(const float* __restrict__ z1, const float* __restrict__ z2,
                       unsigned short* __restrict__ feat, int G) {
  int i = blockIdx.x * 256 + threadIdx.x;
  if (i >= G * 256) return;
  int r = i >> 8, c = i & 255;
  float a = z1[i], b = z2[i];
  size_t base = (size_t)r * 1024;
  feat[base + c]        = f2bf(a);
  feat[base + 256 + c]  = f2bf(b);
  feat[base + 512 + c]  = f2bf(fabsf(a - b));
  feat[base + 768 + c]  = f2bf(a * b);
}

__global__ void k_card2(const float* __restrict__ c1, const float* __restrict__ w2,
                        const float* __restrict__ b2, float* __restrict__ out) {
  __shared__ float red[128];
  int g = blockIdx.x, t = threadIdx.x;
  red[t] = c1[(size_t)g * 128 + t] * w2[t]; __syncthreads();
  for (int s = 64; s > 0; s >>= 1) { if (t < s) red[t] += red[t + s]; __syncthreads(); }
  if (t == 0) {
    float x = red[0] + b2[0];
    out[g] = (x > 20.0f) ? x : log1pf(__expf(x));
  }
}

// ---------------------------------------------------------------------------
extern "C" void kernel_launch(void* const* d_in, const int* in_sizes, int n_in,
                              void* d_out, int out_size, void* d_ws, size_t ws_size,
                              hipStream_t stream) {
  (void)n_in; (void)out_size; (void)ws_size;
  const int Nn = in_sizes[0] / 64;    // nodes
  const int E  = in_sizes[2] / 16;    // edges
  const int G  = in_sizes[8] / 128;   // graphs

  const float* x1 = (const float*)d_in[0];
  const int*  ei1 = (const int*)d_in[1];
  const float* ea1 = (const float*)d_in[2];
  const int*  batch1 = (const int*)d_in[3];
  const float* x2 = (const float*)d_in[4];
  const int*  ei2 = (const int*)d_in[5];
  const float* ea2 = (const float*)d_in[6];
  const int*  batch2 = (const int*)d_in[7];
  const float* aux1 = (const float*)d_in[8];
  const float* aux2 = (const float*)d_in[9];
  auto P = [&](int i) { return (const float*)d_in[i]; };

  // ---- workspace layout ----
  char* basep = (char*)d_ws; size_t off = 0;
  auto alloc = [&](size_t bytes) -> void* {
    off = (off + 255) & ~(size_t)255;
    void* p = basep + off; off += bytes; return p;
  };
  auto allocS = [&](size_t elems) { return (unsigned short*)alloc(elems * 2); };
  auto allocF = [&](size_t elems) { return (float*)alloc(elems * 4); };

  // bf16 weights (converted every launch; K padded to mult of 32)
  unsigned short* w_lin  = allocS(256 * 64);
  unsigned short *w_m1[4], *w_m2[4], *w_ih[4], *w_hh[4];
  for (int l = 0; l < 4; ++l) {
    w_m1[l] = allocS(256 * 288);  w_m2[l] = allocS(256 * 256);
    w_ih[l] = allocS(768 * 256);  w_hh[l] = allocS(768 * 256);
  }
  unsigned short* w_ro   = allocS(256 * 512);
  unsigned short* w_aux1 = allocS(256 * 128);
  unsigned short* w_aux2 = allocS(256 * 256);
  unsigned short* w_fus1 = allocS(256 * 512);
  unsigned short* w_fus2 = allocS(256 * 256);
  unsigned short* w_g12a = allocS(256 * 512);
  unsigned short* w_g12b = allocS(256 * 256);
  unsigned short* w_g21a = allocS(256 * 512);
  unsigned short* w_g21b = allocS(256 * 256);
  unsigned short* w_lab  = allocS(1024 * 256);
  unsigned short* w_sh1  = allocS(512 * 1024);
  unsigned short* w_sh2  = allocS(256 * 512);
  unsigned short* w_cd1  = allocS(128 * 256);

  // activations
  unsigned short* xbf   = allocS((size_t)Nn * 64);
  float*          hbuf  = allocF((size_t)Nn * 256);
  unsigned short* hb    = allocS((size_t)Nn * 256);
  unsigned short* mid   = allocS((size_t)E  * 256);
  float*          agg   = allocF((size_t)Nn * 256);
  unsigned short* aggbf = allocS((size_t)Nn * 256);
  float*          gi    = allocF((size_t)Nn * 768);
  float*          gh    = allocF((size_t)Nn * 768);
  float* zsum = allocF((size_t)G * 256);
  float* zmax = allocF((size_t)G * 256);
  float* cnt  = allocF((size_t)G);
  unsigned short* zcat  = allocS((size_t)G * 512);
  float* z1g = allocF((size_t)G * 256);
  float* z2g = allocF((size_t)G * 256);
  unsigned short* auxbf = allocS((size_t)G * 128);
  float* z1a = allocF((size_t)G * 256);
  float* z2a = allocF((size_t)G * 256);
  unsigned short* cat_b  = allocS((size_t)G * 512);
  unsigned short* gmid_b = allocS((size_t)G * 256);
  float* gbuf = allocF((size_t)G * 256);
  float* pre  = allocF((size_t)G * 256);
  float* z1b  = allocF((size_t)G * 256);
  float* z2b  = allocF((size_t)G * 256);
  float* z1n  = allocF((size_t)G * 256);
  float* z2n  = allocF((size_t)G * 256);
  unsigned short* feat_b = allocS((size_t)G * 1024);
  unsigned short* s1_b   = allocS((size_t)G * 512);
  unsigned short* hh_b   = allocS((size_t)G * 256);
  float* c1out = allocF((size_t)G * 128);

  auto cvtW = [&](const float* src, unsigned short* dst, int O, int I, int Ip) {
    int total = O * Ip;
    k_cvt_pad<<<(total + 255) / 256, 256, 0, stream>>>(src, dst, I, Ip, total);
  };
  auto gemm = [&](const unsigned short* A, const unsigned short* Wt, const float* bias,
                  float* Cf, unsigned short* Cb, float* Cscat, const int* scat,
                  const int* gsrc, const unsigned short* Hs, const float* EA,
                  int M, int N, int K, int act) {
    dim3 grid((unsigned)((M + TM - 1) / TM), (unsigned)(N / TN));
    k_gemm<<<grid, 256, 0, stream>>>(A, Wt, bias, Cf, Cb, Cscat, scat, gsrc, Hs, EA,
                                     M, N, K, act);
  };

  // ---- weight conversion (pytree sorted-key flatten; params begin at d_in[11]) ----
  cvtW(P(79), w_lin, 256, 64, 64);
  for (int l = 0; l < 4; ++l) {
    int bi = 39 + 10 * l;  // Whh, Wih, bhh, bih, bn_b, bn_g, m1W, m1b, m2W, m2b
    cvtW(P(bi + 0), w_hh[l], 768, 256, 256);
    cvtW(P(bi + 1), w_ih[l], 768, 256, 256);
    cvtW(P(bi + 6), w_m1[l], 256, 272, 288);
    cvtW(P(bi + 8), w_m2[l], 256, 256, 256);
  }
  cvtW(P(81), w_ro, 256, 512, 512);
  cvtW(P(11), w_aux1, 256, 128, 128);
  cvtW(P(13), w_aux2, 256, 256, 256);
  cvtW(P(23), w_fus1, 256, 512, 512);
  cvtW(P(25), w_fus2, 256, 256, 256);
  cvtW(P(29), w_g12a, 256, 512, 512);
  cvtW(P(31), w_g12b, 256, 256, 256);
  cvtW(P(33), w_g21a, 256, 512, 512);
  cvtW(P(35), w_g21b, 256, 256, 256);
  cvtW(P(37), w_lab, 1024, 256, 256);
  cvtW(P(83), w_sh1, 512, 1024, 1024);
  cvtW(P(85), w_sh2, 256, 512, 512);
  cvtW(P(15), w_cd1, 128, 256, 256);

  // ---- graph encoder (run twice) ----
  auto encode = [&](const float* x, const int* ei, const float* ea, const int* batch,
                    float* zg) {
    const int* src = ei; const int* dst = ei + E;
    k_cvt<<<(unsigned)(((size_t)Nn * 64 + 255) / 256), 256, 0, stream>>>(x, xbf, (size_t)Nn * 64);
    gemm(xbf, w_lin, P(80), hbuf, hb, nullptr, nullptr, nullptr, nullptr, nullptr,
         Nn, 256, 64, 1);
    for (int l = 0; l < 4; ++l) {
      int bi = 39 + 10 * l;
      // msg1: gather [h[src] | edge_attr | 0], relu -> mid (bf16)
      gemm(nullptr, w_m1[l], P(bi + 7), nullptr, mid, nullptr, nullptr, src, hb, ea,
           E, 256, 288, 1);
      k_zero<<<(unsigned)(((size_t)Nn * 256 + 255) / 256), 256, 0, stream>>>(agg, (size_t)Nn * 256);
      // msg2: scatter-add messages into agg (global_atomic_add_f32)
      gemm(mid, w_m2[l], P(bi + 9), nullptr, nullptr, agg, dst, nullptr, nullptr, nullptr,
           E, 256, 256, 0);
      k_cvt<<<(unsigned)(((size_t)Nn * 256 + 255) / 256), 256, 0, stream>>>(agg, aggbf, (size_t)Nn * 256);
      gemm(aggbf, w_ih[l], P(bi + 3), gi, nullptr, nullptr, nullptr, nullptr, nullptr, nullptr,
           Nn, 768, 256, 0);
      gemm(hb, w_hh[l], P(bi + 2), gh, nullptr, nullptr, nullptr, nullptr, nullptr, nullptr,
           Nn, 768, 256, 0);
      k_gru<<<Nn, 256, 0, stream>>>(gi, gh, hbuf, hb, P(bi + 5), P(bi + 4));
    }
    k_pool_init<<<(G * 256 + 255) / 256, 256, 0, stream>>>(zsum, zmax, cnt, G);
    k_pool<<<Nn, 256, 0, stream>>>(hbuf, batch, zsum, zmax, cnt);
    k_pool_final<<<(G * 256 + 255) / 256, 256, 0, stream>>>(zsum, zmax, cnt, zcat, G);
    gemm(zcat, w_ro, P(82), zg, nullptr, nullptr, nullptr, nullptr, nullptr, nullptr,
         G, 256, 512, 1);
  };
  encode(x1, ei1, ea1, batch1, z1g);
  encode(x2, ei2, ea2, batch2, z2g);

  // ---- aux MLPs ----
  k_cvt<<<(G * 128 + 255) / 256, 256, 0, stream>>>(aux1, auxbf, (size_t)G * 128);
  gemm(auxbf, w_aux1, P(12), nullptr, gmid_b, nullptr, nullptr, nullptr, nullptr, nullptr, G, 256, 128, 1);
  gemm(gmid_b, w_aux2, P(14), z1a, nullptr, nullptr, nullptr, nullptr, nullptr, nullptr, G, 256, 256, 1);
  k_cvt<<<(G * 128 + 255) / 256, 256, 0, stream>>>(aux2, auxbf, (size_t)G * 128);
  gemm(auxbf, w_aux1, P(12), nullptr, gmid_b, nullptr, nullptr, nullptr, nullptr, nullptr, G, 256, 128, 1);
  gemm(gmid_b, w_aux2, P(14), z2a, nullptr, nullptr, nullptr, nullptr, nullptr, nullptr, G, 256, 256, 1);

  const int GE = G * 256, GB = (GE + 255) / 256;
  // ---- fusion gates + LN ----
  k_concat2<<<(G * 512 + 255) / 256, 256, 0, stream>>>(z1g, z1a, cat_b, 256, 256, G);
  gemm(cat_b, w_fus1, P(24), nullptr, gmid_b, nullptr, nullptr, nullptr, nullptr, nullptr, G, 256, 512, 1);
  gemm(gmid_b, w_fus2, P(26), gbuf, nullptr, nullptr, nullptr, nullptr, nullptr, nullptr, G, 256, 256, 2);
  k_mix<<<GB, 256, 0, stream>>>(gbuf, z1g, z1a, pre, GE, 0);
  k_ln<<<G, 256, 0, stream>>>(pre, P(28), P(27), z1b);
  k_concat2<<<(G * 512 + 255) / 256, 256, 0, stream>>>(z2g, z2a, cat_b, 256, 256, G);
  gemm(cat_b, w_fus1, P(24), nullptr, gmid_b, nullptr, nullptr, nullptr, nullptr, nullptr, G, 256, 512, 1);
  gemm(gmid_b, w_fus2, P(26), gbuf, nullptr, nullptr, nullptr, nullptr, nullptr, nullptr, G, 256, 256, 2);
  k_mix<<<GB, 256, 0, stream>>>(gbuf, z2g, z2a, pre, GE, 0);
  k_ln<<<G, 256, 0, stream>>>(pre, P(28), P(27), z2b);

  // ---- cross gates + LN ----
  k_concat2<<<(G * 512 + 255) / 256, 256, 0, stream>>>(z1b, z2b, cat_b, 256, 256, G);
  gemm(cat_b, w_g12a, P(30), nullptr, gmid_b, nullptr, nullptr, nullptr, nullptr, nullptr, G, 256, 512, 1);
  gemm(gmid_b, w_g12b, P(32), gbuf, nullptr, nullptr, nullptr, nullptr, nullptr, nullptr, G, 256, 256, 2);
  k_mix<<<GB, 256, 0, stream>>>(gbuf, z1b, z2b, pre, GE, 1);
  k_ln<<<G, 256, 0, stream>>>(pre, P(20), P(19), z1n);
  k_concat2<<<(G * 512 + 255) / 256, 256, 0, stream>>>(z2b, z1b, cat_b, 256, 256, G);
  gemm(cat_b, w_g21a, P(34), nullptr, gmid_b, nullptr, nullptr, nullptr, nullptr, nullptr, G, 256, 512, 1);
  gemm(gmid_b, w_g21b, P(36), gbuf, nullptr, nullptr, nullptr, nullptr, nullptr, nullptr, G, 256, 256, 2);
  k_mix<<<GB, 256, 0, stream>>>(gbuf, z2b, z1b, pre, GE, 1);
  k_ln<<<G, 256, 0, stream>>>(pre, P(22), P(21), z2n);

  // ---- shared head ----
  k_feat<<<GB, 256, 0, stream>>>(z1n, z2n, feat_b, G);
  gemm(feat_b, w_sh1, P(84), nullptr, s1_b, nullptr, nullptr, nullptr, nullptr, nullptr, G, 512, 1024, 1);
  gemm(s1_b, w_sh2, P(86), nullptr, hh_b, nullptr, nullptr, nullptr, nullptr, nullptr, G, 256, 512, 1);
  // logits -> d_out[0 : G*1024)
  gemm(hh_b, w_lab, P(38), (float*)d_out, nullptr, nullptr, nullptr, nullptr, nullptr, nullptr, G, 1024, 256, 0);
  // cardinality -> d_out[G*1024 : G*1024+G)
  gemm(hh_b, w_cd1, P(16), c1out, nullptr, nullptr, nullptr, nullptr, nullptr, nullptr, G, 128, 256, 1);
  k_card2<<<G, 128, 0, stream>>>(c1out, P(17), P(18), (float*)d_out + (size_t)G * 1024);
}
